// CL_76493367542060
// MI455X (gfx1250) — matrix operand order repaired and checked by reading, not compiled, not run on hardware
//
#include <hip/hip_runtime.h>
#include <hip/hip_bf16.h>
#include <math.h>

#define CN 4096
#define CD 1024
#define NCLS 128
#define NS (CD / 32)          // 32 k-steps of K=32
#define LPAD 40               // padded LDS row stride in halves (80B = 20 dwords)

typedef __attribute__((ext_vector_type(16))) _Float16 v16h;
typedef __attribute__((ext_vector_type(8)))  float    v8f;

union FragU { v16h v; float4 f[2]; };

// ---- CDNA5 async global->LDS copy (ASYNCcnt), ISA 15.18.3 op 98 ----
// VDST = per-lane LDS byte offset, VADDR = 64-bit global address.
#define CL_ASYNC_CP16(ldsoff, gaddr)                                   \
    asm volatile("global_load_async_to_lds_b128 %0, %1, off"           \
                 :: "v"(ldsoff), "v"(gaddr) : "memory")

#if __has_builtin(__builtin_amdgcn_s_wait_asynccnt)
#define CL_WAIT_ASYNC(n) __builtin_amdgcn_s_wait_asynccnt(n)
#else
#define CL_WAIT_ASYNC(n) asm volatile("s_wait_asynccnt %0" :: "i"(n) : "memory")
#endif

static __device__ __forceinline__ unsigned cl_lds_off(const void* p) {
    // AMDGPU generic->LDS: offset lives in the low 32 bits of the flat address.
    return (unsigned)(uintptr_t)p;
}

__global__ void cl_init_kernel(float* __restrict__ rowsum, int* __restrict__ cc,
                               float* __restrict__ out) {
    int i = blockIdx.x * blockDim.x + threadIdx.x;
    if (i < CN) rowsum[i] = 0.0f;
    if (i < NCLS) cc[i] = 0;
    if (i == 0) out[0] = 0.0f;
}

// One block (256 thr) per row: sum of squares -> f16 normalized row, + histogram.
__global__ void cl_normalize_kernel(const float* __restrict__ X,
                                    const int* __restrict__ y,
                                    _Float16* __restrict__ Xh,
                                    int* __restrict__ cc) {
    const int row = blockIdx.x;
    const int tid = threadIdx.x;
    const int lane = tid & 31, wid = tid >> 5;
    float xv[4];
    float ss = 0.0f;
#pragma unroll
    for (int t = 0; t < 4; ++t) {
        xv[t] = X[(size_t)row * CD + tid + t * 256];
        ss += xv[t] * xv[t];
    }
#pragma unroll
    for (int m = 1; m < 32; m <<= 1) ss += __shfl_xor(ss, m, 32);
    __shared__ float sred[8];
    if (lane == 0) sred[wid] = ss;
    __syncthreads();
    float tot = 0.0f;
#pragma unroll
    for (int w = 0; w < 8; ++w) tot += sred[w];
    float scale = 1.0f / fmaxf(sqrtf(tot), 1e-8f);
#pragma unroll
    for (int t = 0; t < 4; ++t)
        Xh[(size_t)row * CD + tid + t * 256] = (_Float16)(xv[t] * scale);
    if (tid == 0) atomicAdd(&cc[y[row]], 1);
}

// first j != i with y[j] == y[i]; -1 if none. Label-only, cheap scan.
__global__ void cl_firstpos_kernel(const int* __restrict__ y, int* __restrict__ fp) {
    int i = blockIdx.x * blockDim.x + threadIdx.x;
    if (i >= CN) return;
    int yi = y[i];
    int r = -1;
    for (int j = 0; j < CN; ++j) {
        if (j != i && y[j] == yi) { r = j; break; }
    }
    fp[i] = r;
}

// Fused Gram (async LDS staging + WMMA f16->f32) + masked exp + row/col scatter.
// Upper-triangular blocks only (by <= bx); off-diagonal blocks also scatter the
// mirror (column) contribution.
__global__ __launch_bounds__(256)
void cl_gram_kernel(const _Float16* __restrict__ Xh,
                    const int* __restrict__ y,
                    float* __restrict__ rowsum) {
    const int bx = blockIdx.x, by = blockIdx.y;
    if (by > bx) return;

    __shared__ _Float16 ldsA[3][128][LPAD];
    __shared__ _Float16 ldsB[3][128][LPAD];

    const int tid = threadIdx.x;
    const int lane = tid & 31, wid = tid >> 5;
    const int wr = wid >> 1, wc = wid & 1;        // 4x2 wave tiling of 128x128
    const int rowW = by * 128 + wr * 32;
    const int colW = bx * 128 + wc * 64;
    const int halfsel = lane >> 4;
    const int q = lane & 15;

    // Copy descriptors: thread tid handles 16B chunks c = tid and c = tid+256
    // of each 128x32-half strip (4 chunks per row).
    const int rA = tid >> 2;                      // 0..63
    const int kf = (tid & 3) * 8;                 // half offset within k-chunk
    const _Float16* gA = Xh + (size_t)(by * 128 + rA) * CD + kf;
    const _Float16* gB = Xh + (size_t)(bx * 128 + rA) * CD + kf;

#define CL_ISSUE_BATCH(buf, k0)                                                      \
    do {                                                                             \
        CL_ASYNC_CP16(cl_lds_off(&ldsA[buf][rA][kf]),                                \
                      (unsigned long long)(uintptr_t)(gA + (k0)));                   \
        CL_ASYNC_CP16(cl_lds_off(&ldsA[buf][rA + 64][kf]),                           \
                      (unsigned long long)(uintptr_t)(gA + (size_t)64 * CD + (k0))); \
        CL_ASYNC_CP16(cl_lds_off(&ldsB[buf][rA][kf]),                                \
                      (unsigned long long)(uintptr_t)(gB + (k0)));                   \
        CL_ASYNC_CP16(cl_lds_off(&ldsB[buf][rA + 64][kf]),                           \
                      (unsigned long long)(uintptr_t)(gB + (size_t)64 * CD + (k0))); \
    } while (0)

    v8f zero;
#pragma unroll
    for (int e = 0; e < 8; ++e) zero[e] = 0.0f;
    v8f acc[2][4];
#pragma unroll
    for (int mt = 0; mt < 2; ++mt)
#pragma unroll
        for (int nt = 0; nt < 4; ++nt) acc[mt][nt] = zero;

    // labels for masking (independent of LDS pipeline)
    int yj[4];
#pragma unroll
    for (int nt = 0; nt < 4; ++nt) yj[nt] = y[colW + nt * 16 + q];
    int yi[2][8];
#pragma unroll
    for (int mt = 0; mt < 2; ++mt)
#pragma unroll
        for (int r = 0; r < 8; ++r)
            yi[mt][r] = y[rowW + mt * 16 + r + 8 * halfsel];

    // prologue: prefetch batches 0 and 1
    CL_ISSUE_BATCH(0, 0);
    CL_ISSUE_BATCH(1, 32);

    for (int s = 0; s < NS; ++s) {
        // wait for batch s (4 newer copies of batch s+1 may stay in flight)
        if (s + 1 < NS) CL_WAIT_ASYNC(4); else CL_WAIT_ASYNC(0);
        __syncthreads();                    // all waves' batch-s copies visible;
                                            // also fences reuse of buffer (s+2)%3
        if (s + 2 < NS) {
            const int nb = (s + 2) % 3;
            CL_ISSUE_BATCH(nb, (s + 2) * 32);
        }

        const int buf = s % 3;
        // A fragments: lane = row q of the 16-row tile; K chunks {h*8, 16+h*8}
        v16h afr[2];
#pragma unroll
        for (int mt = 0; mt < 2; ++mt) {
            FragU u;
            u.f[0] = *(const float4*)&ldsA[buf][wr * 32 + mt * 16 + q][halfsel * 8];
            u.f[1] = *(const float4*)&ldsA[buf][wr * 32 + mt * 16 + q][16 + halfsel * 8];
            afr[mt] = u.v;
        }
        // B fragments: lane = column q; K chunks {h*8, 16+h*8}
        v16h bfr[4];
#pragma unroll
        for (int nt = 0; nt < 4; ++nt) {
            FragU u;
            const float4* pb =
                (const float4*)&ldsB[buf][wc * 64 + nt * 16 + q][halfsel * 8];
            u.f[0] = pb[0];
            u.f[1] = pb[2];
            bfr[nt] = u.v;
        }
#pragma unroll
        for (int mt = 0; mt < 2; ++mt)
#pragma unroll
            for (int nt = 0; nt < 4; ++nt)
                acc[mt][nt] = __builtin_amdgcn_wmma_f32_16x16x32_f16(
                    false, afr[mt], false, bfr[nt],
                    (short)0, acc[mt][nt], false, false);
    }

    // overwrite acc with masked exp((cos+1)*0.25)
#pragma unroll
    for (int mt = 0; mt < 2; ++mt)
#pragma unroll
        for (int nt = 0; nt < 4; ++nt)
#pragma unroll
            for (int r = 0; r < 8; ++r) {
                float sv = (acc[mt][nt][r] + 1.0f) * 0.25f;
                acc[mt][nt][r] = (yi[mt][r] != yj[nt]) ? __expf(sv) : 0.0f;
            }

    // row sums: reduce over the 16 lanes of each half (N dimension)
#pragma unroll
    for (int mt = 0; mt < 2; ++mt)
#pragma unroll
        for (int r = 0; r < 8; ++r) {
            float v = acc[mt][0][r] + acc[mt][1][r] + acc[mt][2][r] + acc[mt][3][r];
#pragma unroll
            for (int m = 1; m < 16; m <<= 1) v += __shfl_xor(v, m, 32);
            if (q == 0)
                atomicAdd(&rowsum[rowW + mt * 16 + r + 8 * halfsel], v);
        }

    // column sums (mirror contribution), off-diagonal blocks only
    if (bx != by) {
#pragma unroll
        for (int nt = 0; nt < 4; ++nt) {
            float v = 0.0f;
#pragma unroll
            for (int mt = 0; mt < 2; ++mt)
#pragma unroll
                for (int r = 0; r < 8; ++r) v += acc[mt][nt][r];
            v += __shfl_xor(v, 16, 32);   // combine halves -> full 32-row column sum
            if (halfsel == 0)
                atomicAdd(&rowsum[colW + nt * 16 + q], v);
        }
    }
#undef CL_ISSUE_BATCH
}

// One wave per row: pos_val dot product + closed-form logsumexp + global sum.
__global__ void cl_finalize_kernel(const _Float16* __restrict__ Xh,
                                   const int* __restrict__ y,
                                   const float* __restrict__ rowsum,
                                   const int* __restrict__ fp,
                                   const int* __restrict__ cc,
                                   float* __restrict__ out) {
    const int lane = threadIdx.x & 31, wid = threadIdx.x >> 5;
    const int i = blockIdx.x * 8 + wid;
    if (i >= CN) return;
    float pos = 0.0f;
    int p = fp[i];
    if (p >= 0) {
        float d = 0.0f;
        for (int k = lane; k < CD; k += 32)
            d += (float)Xh[(size_t)i * CD + k] * (float)Xh[(size_t)p * CD + k];
#pragma unroll
        for (int m = 1; m < 32; m <<= 1) d += __shfl_xor(d, m, 32);
        pos = (d + 1.0f) * 0.25f;
    }
    if (lane == 0) {
        int nneg = CN - cc[y[i]];
        float zcount = (float)(2 * CN - 2 - nneg);
        float lse = logf(__expf(pos) + rowsum[i] + zcount);
        atomicAdd(out, (lse - pos) * (1.0f / (float)CN));
    }
}

extern "C" void kernel_launch(void* const* d_in, const int* in_sizes, int n_in,
                              void* d_out, int out_size, void* d_ws, size_t ws_size,
                              hipStream_t stream) {
    const float* X = (const float*)d_in[0];
    const int*   y = (const int*)d_in[1];
    float* out = (float*)d_out;

    char* ws = (char*)d_ws;
    _Float16* Xh   = (_Float16*)ws;                                   // 8 MB
    float* rowsum  = (float*)(ws + (size_t)CN * CD * 2);              // 16 KB
    int*   fp      = (int*)(ws + (size_t)CN * CD * 2 + CN * 4);       // 16 KB
    int*   cc      = (int*)(ws + (size_t)CN * CD * 2 + CN * 8);       // 512 B

    cl_init_kernel<<<16, 256, 0, stream>>>(rowsum, cc, out);
    cl_normalize_kernel<<<CN, 256, 0, stream>>>(X, y, Xh, cc);
    cl_firstpos_kernel<<<16, 256, 0, stream>>>(y, fp);
    cl_gram_kernel<<<dim3(32, 32), 256, 0, stream>>>(Xh, y, rowsum);
    cl_finalize_kernel<<<512, 256, 0, stream>>>(Xh, y, rowsum, fp, cc, out);
}
